// QKVMultiheadAttention_56538949484962
// MI455X (gfx1250) — matrix operand order repaired and checked by reading, not compiled, not run on hardware
//
#include <hip/hip_runtime.h>

// ---- CDNA5 wave32 WMMA types ----
typedef __attribute__((ext_vector_type(16))) _Float16 v16h;
typedef __attribute__((ext_vector_type(8)))  _Float16 h8;
typedef __attribute__((ext_vector_type(4)))  _Float16 hv4;
typedef __attribute__((ext_vector_type(8)))  float    v8f;

#define CTX   2048
#define WIDTH 1024
#define NH    16
#define DH    64      // head dim
#define QT    64      // q rows per workgroup
#define KBLK  32      // keys per iteration
#define NWAVE 4       // 128 threads

// Q is pre-scaled by (1/8)*log2(e); K by 1/8  -> S is already in exp2 domain
#define QSCALE 0.18033688f   // 0.125f * 1.44269504f
#define KSCALE 0.125f

static __device__ __forceinline__ float fast_exp2(float x) {
  return __builtin_amdgcn_exp2f(x);   // v_exp_f32
}

static __device__ __forceinline__ v16h cat16(h8 lo, h8 hi) {
  return __builtin_shufflevector(lo, hi, 0,1,2,3,4,5,6,7,8,9,10,11,12,13,14,15);
}

static __device__ __forceinline__ v8f wmma_f16(v16h a, v16h b, v8f c) {
  // D = A(16x32 f16) * B(32x16 f16) + C(16x16 f32)
  return __builtin_amdgcn_wmma_f32_16x16x32_f16(
      /*neg_a=*/false, a, /*neg_b=*/false, b,
      /*c_mod=*/(short)0, c, /*reuse_a=*/false, /*reuse_b=*/false);
}

__launch_bounds__(128, 1)
__global__ void fa_fwd_kernel(const float* __restrict__ Q,
                              const float* __restrict__ K,
                              const float* __restrict__ V,
                              float* __restrict__ O)
{
  // LDS staging (20 KB total; WGP has 320 KB)
  __shared__ __align__(16) _Float16 Qs[QT * DH];            // 8 KB, [qrow][c], scaled
  __shared__ __align__(16) _Float16 Ks[KBLK * DH];          // 4 KB, [key][c],  scaled
  __shared__ __align__(16) _Float16 Vt[DH * KBLK];          // 4 KB, [c][key]  (transposed)
  __shared__ __align__(16) _Float16 Ps[NWAVE][16 * KBLK];   // 4 KB, per-wave P tile [m][s]

  const int tid  = threadIdx.x;
  const int wave = tid >> 5;
  const int lane = tid & 31;
  const int hlf  = lane >> 4;      // 0: lanes 0-15, 1: lanes 16-31
  const int l16  = lane & 15;

  const int qtile = blockIdx.x % (CTX / QT);
  const int head  = (blockIdx.x / (CTX / QT)) % NH;
  const int batch =  blockIdx.x / ((CTX / QT) * NH);
  const int q0    = qtile * QT;

  // (b,h) panel base: row r is at base + r*WIDTH
  const size_t base = (size_t)batch * CTX * WIDTH + (size_t)head * DH;

  // ---- stage Q tile (64x64 fp32 -> f16 * QSCALE), 1024 float4 chunks / 128 thr ----
#pragma unroll
  for (int i = 0; i < 8; ++i) {
    int chunk = tid + i * 128;
    int row   = chunk >> 4;
    int c4    = (chunk & 15) << 2;
    const float4 f = *(const float4*)(Q + base + (size_t)(q0 + row) * WIDTH + c4);
    hv4 hq;
    hq.x = (_Float16)(f.x * QSCALE); hq.y = (_Float16)(f.y * QSCALE);
    hq.z = (_Float16)(f.z * QSCALE); hq.w = (_Float16)(f.w * QSCALE);
    *(hv4*)&Qs[row * DH + c4] = hq;
  }
  __syncthreads();

  // ---- build Q A-fragments: wave's rows m = wave*16 + l16, two 32-chan chunks ----
  // A 16x32 f16, k order chosen here (B matches): lane m holds K = e + 16*hlf
  v16h aq[2];
  {
    const int m = wave * 16 + l16;
#pragma unroll
    for (int f = 0; f < 2; ++f) {
      h8 lo = *(const h8*)&Qs[m * DH + f * 32 + 16 * hlf];
      h8 hi = *(const h8*)&Qs[m * DH + f * 32 + 16 * hlf + 8];
      aq[f] = cat16(lo, hi);
    }
  }

  // all-ones B fragment for row-sum-by-WMMA
  v16h ones;
#pragma unroll
  for (int e = 0; e < 16; ++e) ones[e] = (_Float16)1.0f;

  // online-softmax state (exp2 domain); lane group `hlf` owns rows M = 8*hlf + r
  float rm[8], rl[8];
  v8f oacc[4];
#pragma unroll
  for (int r = 0; r < 8; ++r) { rm[r] = -3.0e38f; rl[r] = 0.0f; }
#pragma unroll
  for (int g = 0; g < 4; ++g)
#pragma unroll
    for (int r = 0; r < 8; ++r) oacc[g][r] = 0.0f;

  const size_t kvstep = (size_t)KBLK * WIDTH;

  for (int kb = 0; kb < CTX / KBLK; ++kb) {
    __syncthreads();   // protect prior-iteration LDS readers

    // ---- stage K (f16 * 1/8) and V (f16, transposed): 512 float4 chunks each ----
#pragma unroll
    for (int i = 0; i < 4; ++i) {
      int chunk = tid + i * 128;
      int row   = chunk >> 4;
      int c4    = (chunk & 15) << 2;
      size_t g  = base + (size_t)(kb * KBLK + row) * WIDTH + c4;
      float4 fk = *(const float4*)(K + g);
      float4 fv = *(const float4*)(V + g);
      hv4 hk;
      hk.x = (_Float16)(fk.x * KSCALE); hk.y = (_Float16)(fk.y * KSCALE);
      hk.z = (_Float16)(fk.z * KSCALE); hk.w = (_Float16)(fk.w * KSCALE);
      *(hv4*)&Ks[row * DH + c4] = hk;
      Vt[(c4 + 0) * KBLK + row] = (_Float16)fv.x;
      Vt[(c4 + 1) * KBLK + row] = (_Float16)fv.y;
      Vt[(c4 + 2) * KBLK + row] = (_Float16)fv.z;
      Vt[(c4 + 3) * KBLK + row] = (_Float16)fv.w;
      if (kb + 1 < CTX / KBLK) {
        __builtin_prefetch(K + g + kvstep, 0, 1);   // global_prefetch_b8
        __builtin_prefetch(V + g + kvstep, 0, 1);
      }
    }
    __syncthreads();

    // ---- S = Q' K'^T : two 16-key tiles, K-dim 64 = 2 chained WMMAs each ----
    // B 32x16, k order chosen here (same as A): lane n = key, element e -> kdim = e + 16*hlf
    v8f s0, s1;
#pragma unroll
    for (int r = 0; r < 8; ++r) { s0[r] = 0.0f; s1[r] = 0.0f; }
#pragma unroll
    for (int f = 0; f < 2; ++f) {
      h8 lo0 = *(const h8*)&Ks[l16 * DH        + f * 32 + 16 * hlf];
      h8 hi0 = *(const h8*)&Ks[l16 * DH        + f * 32 + 16 * hlf + 8];
      h8 lo1 = *(const h8*)&Ks[(16 + l16) * DH + f * 32 + 16 * hlf];
      h8 hi1 = *(const h8*)&Ks[(16 + l16) * DH + f * 32 + 16 * hlf + 8];
      s0 = wmma_f16(aq[f], cat16(lo0, hi0), s0);
      s1 = wmma_f16(aq[f], cat16(lo1, hi1), s1);
    }

    // ---- row max: batched cross-lane reduce within each 16-lane group ----
    float t[8];
#pragma unroll
    for (int r = 0; r < 8; ++r) t[r] = fmaxf(s0[r], s1[r]);
#pragma unroll
    for (int msk = 1; msk <= 8; msk <<= 1) {
      float sh[8];
#pragma unroll
      for (int r = 0; r < 8; ++r) sh[r] = __shfl_xor(t[r], msk, 32);
#pragma unroll
      for (int r = 0; r < 8; ++r) t[r] = fmaxf(t[r], sh[r]);
    }

    // ---- rescale state, exponentiate, spill P tile to LDS (C->A layout) ----
    float alpha[8];
#pragma unroll
    for (int r = 0; r < 8; ++r) {
      float mnew = fmaxf(rm[r], t[r]);
      alpha[r]   = fast_exp2(rm[r] - mnew);   // exp2 domain already
      rm[r]      = mnew;
      float p0   = fast_exp2(s0[r] - mnew);
      float p1   = fast_exp2(s1[r] - mnew);
      Ps[wave][(r + 8 * hlf) * KBLK + l16]      = (_Float16)p0;
      Ps[wave][(r + 8 * hlf) * KBLK + 16 + l16] = (_Float16)p1;
#pragma unroll
      for (int g = 0; g < 4; ++g) oacc[g][r] *= alpha[r];
    }
    __syncthreads();

    // ---- P A-fragment (16x32, K-dim = 32 keys) ----
    v16h ap;
    {
      h8 lo = *(const h8*)&Ps[wave][l16 * KBLK + 16 * hlf];
      h8 hi = *(const h8*)&Ps[wave][l16 * KBLK + 16 * hlf + 8];
      ap = cat16(lo, hi);
    }

    // ---- row sums via WMMA against all-ones B: rs[r] replicated across group ----
    v8f zc;
#pragma unroll
    for (int r = 0; r < 8; ++r) zc[r] = 0.0f;
    v8f rs = wmma_f16(ap, ones, zc);
#pragma unroll
    for (int r = 0; r < 8; ++r) rl[r] = rl[r] * alpha[r] + rs[r];

    // ---- O += P * V : 4 channel groups of 16 ----
#pragma unroll
    for (int g = 0; g < 4; ++g) {
      h8 lo = *(const h8*)&Vt[(g * 16 + l16) * KBLK + 16 * hlf];
      h8 hi = *(const h8*)&Vt[(g * 16 + l16) * KBLK + 16 * hlf + 8];
      oacc[g] = wmma_f16(ap, cat16(lo, hi), oacc[g]);
    }
  }

  // ---- normalize and store fp32 output ----
#pragma unroll
  for (int r = 0; r < 8; ++r) {
    float inv = 1.0f / rl[r];
    int qrow  = q0 + wave * 16 + r + 8 * hlf;
    size_t rowbase = base + (size_t)qrow * WIDTH;
#pragma unroll
    for (int g = 0; g < 4; ++g) {
      O[rowbase + g * 16 + l16] = oacc[g][r] * inv;
    }
  }
}

extern "C" void kernel_launch(void* const* d_in, const int* in_sizes, int n_in,
                              void* d_out, int out_size, void* d_ws, size_t ws_size,
                              hipStream_t stream) {
  (void)n_in; (void)d_ws; (void)ws_size; (void)out_size;
  const float* q = (const float*)d_in[0];
  const float* k = (const float*)d_in[1];
  const float* v = (const float*)d_in[2];
  float* o = (float*)d_out;
  const int bs = in_sizes[0] / (CTX * WIDTH);
  dim3 grid(bs * NH * (CTX / QT));
  fa_fwd_kernel<<<grid, 128, 0, stream>>>(q, k, v, o);
}
